// GRP_17437567222434
// MI455X (gfx1250) — matrix operand order repaired and checked by reading, not compiled, not run on hardware
//
#include <hip/hip_runtime.h>

// ---------------- CDNA5 WMMA types ----------------
typedef __attribute__((ext_vector_type(16))) __bf16 v16bf;
typedef __attribute__((ext_vector_type(8)))  float  v8f;

#define D_   768
#define SEQ_ 667
#define B_   8
#define H_   12
#define HD_  64
#define MTOK (B_*SEQ_)   /* 5336 */
#define FF_  3072
#define EPS_ 1e-5f

__device__ __forceinline__ __bf16 f2bf(float x) { return (__bf16)x; }

__device__ __forceinline__ v8f wmma_bf16(v16bf a, v16bf b, v8f c) {
  // D = A(16x32) * B(32x16) + C, f32 accumulate
  return __builtin_amdgcn_wmma_f32_16x16x32_bf16(false, a, false, b, (short)0, c, false, false);
}

// ---------------- operand packing into WMMA fragment layout ----------------
// A tile (16x32 bf16): lanes 0-15 row M=lane, elems = K0..15;
// lanes 16-31 row M=lane-16, elems = K16..31.
__global__ __launch_bounds__(32) void pack_a_kernel(const float* __restrict__ A,
                                                    __bf16* __restrict__ Ap,
                                                    int Mreal, int K) {
  int kt = blockIdx.x, mt = blockIdx.y, lane = threadIdx.x;
  int Ktiles = gridDim.x;
  int m = mt * 16 + (lane & 15); if (m >= Mreal) m = Mreal - 1;
  int koff = kt * 32 + ((lane >> 4) << 4);
  const float* ar = A + (size_t)m * K + koff;
  v16bf o;
#pragma unroll
  for (int j = 0; j < 8; ++j) { o[j] = f2bf(ar[j]); o[8 + j] = f2bf(ar[8 + j]); }
  *(v16bf*)(Ap + ((size_t)mt * Ktiles + kt) * 512 + lane * 16) = o;
}

// B tile (32x16 bf16): lanes 0-15 col N=lane, elems = K0..15;
// lanes 16-31 col N=lane-16, elems = K16..31.
__global__ __launch_bounds__(32) void pack_b_kernel(const float* __restrict__ W,
                                                    __bf16* __restrict__ Bp,
                                                    int K, int N) {
  int nt = blockIdx.x, kt = blockIdx.y, lane = threadIdx.x;
  int Ntiles = gridDim.x;
  int n  = nt * 16 + (lane & 15);
  int kb = kt * 32 + ((lane < 16) ? 0 : 16);
  v16bf o;
#pragma unroll
  for (int e = 0; e < 16; ++e) o[e] = f2bf(W[(size_t)(kb + e) * N + n]);
  *(v16bf*)(Bp + ((size_t)kt * Ntiles + nt) * 512 + lane * 16) = o;
}

// ---------------- WMMA GEMM ----------------
// Workgroup = 4 waves = 128x64 output tile. Each wave computes 32x64 (2 M-tiles,
// 8 WMMA per K-tile). The 4KB B-panel per K-tile is shared by all 4 waves:
// fetched with CDNA5 async loads (global_load_async_to_lds_b128, ASYNCcnt),
// double-buffered in LDS, prefetched one K-tile ahead. All 4 B fragments are
// loaded into distinct registers before the 8 back-to-back WMMAs so the XDL
// pipe is not stalled on ds waits between pairs.
// Output row remap: orow = (m/rpb)*seqStride + tokOff + m%rpb (identity: rpb=M,0,0)
__global__ __launch_bounds__(128) void gemm_wmma_kernel(const __bf16* __restrict__ Ap,
                                                        const __bf16* __restrict__ Bp,
                                                        const float* __restrict__ bias,
                                                        const float* __restrict__ resid,
                                                        float* __restrict__ outp,
                                                        int Mreal, int N, int Ktiles,
                                                        int rpb, int tokOff, int seqStride,
                                                        int doRelu) {
  __shared__ __bf16 bsh[2][2048];          // 2 x 4KB B panel (4 n-tiles)
  int nb = blockIdx.x, mblk = blockIdx.y;
  int tid = threadIdx.x;
  int wave = tid >> 5, lane = tid & 31;
  int Ntiles = N >> 4;

  int mt0 = mblk * 8 + wave * 2;           // this wave's two M-tiles
  const __bf16* apT0 = Ap + ((size_t)mt0 * Ktiles) * 512 + lane * 16;
  const __bf16* apT1 = apT0 + (size_t)Ktiles * 512;
  const __bf16* bg   = Bp + ((size_t)nb * 4) * 512;   // + kt*Ntiles*512

  // async-stage one K-tile's 4KB B panel into LDS buffer `buf`
  auto stage = [&](int kt, int buf) {
    unsigned long long g0 = (unsigned long long)(uintptr_t)(bg + (size_t)kt * Ntiles * 512)
                          + (unsigned long long)tid * 16;
    unsigned l0 = (unsigned)(uintptr_t)(&bsh[buf][0]) + (unsigned)tid * 16;
    asm volatile("global_load_async_to_lds_b128 %0, %1, off"
                 :: "v"(l0), "v"(g0) : "memory");
    asm volatile("global_load_async_to_lds_b128 %0, %1, off"
                 :: "v"(l0 + 2048u), "v"(g0 + 2048ull) : "memory");
  };

  const v8f Z8 = {0.f,0.f,0.f,0.f,0.f,0.f,0.f,0.f};
  v8f acc[2][4];
#pragma unroll
  for (int h = 0; h < 2; ++h)
#pragma unroll
    for (int t = 0; t < 4; ++t) acc[h][t] = Z8;

  stage(0, 0);
  for (int kt = 0; kt < Ktiles; ++kt) {
    if (kt + 1 < Ktiles) {
      stage(kt + 1, (kt + 1) & 1);
      asm volatile("s_wait_asynccnt 0x2" ::: "memory");  // current tile's 2 done
    } else {
      asm volatile("s_wait_asynccnt 0x0" ::: "memory");
    }
    __syncthreads();                                     // LDS visible to all waves
    v16bf a0 = *(const v16bf*)(apT0 + (size_t)kt * 512);
    v16bf a1 = *(const v16bf*)(apT1 + (size_t)kt * 512);
    const __bf16* bs = &bsh[kt & 1][lane * 16];
    // load all 4 B fragments first (distinct regs), then 8 back-to-back WMMAs
    v16bf b0 = *(const v16bf*)(bs);
    v16bf b1 = *(const v16bf*)(bs + 512);
    v16bf b2 = *(const v16bf*)(bs + 1024);
    v16bf b3 = *(const v16bf*)(bs + 1536);
    acc[0][0] = wmma_bf16(a0, b0, acc[0][0]);
    acc[1][0] = wmma_bf16(a1, b0, acc[1][0]);
    acc[0][1] = wmma_bf16(a0, b1, acc[0][1]);
    acc[1][1] = wmma_bf16(a1, b1, acc[1][1]);
    acc[0][2] = wmma_bf16(a0, b2, acc[0][2]);
    acc[1][2] = wmma_bf16(a1, b2, acc[1][2]);
    acc[0][3] = wmma_bf16(a0, b3, acc[0][3]);
    acc[1][3] = wmma_bf16(a1, b3, acc[1][3]);
    __syncthreads();                                     // done with this buffer
  }

  int n0 = nb * 64 + (lane & 15);
  float bv0 = bias ? bias[n0]      : 0.f;
  float bv1 = bias ? bias[n0 + 16] : 0.f;
  float bv2 = bias ? bias[n0 + 32] : 0.f;
  float bv3 = bias ? bias[n0 + 48] : 0.f;
#pragma unroll
  for (int h = 0; h < 2; ++h) {
    int mb = (mt0 + h) * 16 + ((lane < 16) ? 0 : 8);
#pragma unroll
    for (int r = 0; r < 8; ++r) {
      int m = mb + r;
      if (m < Mreal) {
        int orow = (m / rpb) * seqStride + tokOff + (m % rpb);
        size_t base = (size_t)orow * N;
        float v0 = acc[h][0][r] + bv0, v1 = acc[h][1][r] + bv1;
        float v2 = acc[h][2][r] + bv2, v3 = acc[h][3][r] + bv3;
        if (doRelu) { v0 = fmaxf(v0, 0.f); v1 = fmaxf(v1, 0.f);
                      v2 = fmaxf(v2, 0.f); v3 = fmaxf(v3, 0.f); }
        if (resid)  { v0 += resid[base + n0]; v1 += resid[base + n0 + 16];
                      v2 += resid[base + n0 + 32]; v3 += resid[base + n0 + 48]; }
        outp[base + n0]      = v0;
        outp[base + n0 + 16] = v1;
        outp[base + n0 + 32] = v2;
        outp[base + n0 + 48] = v3;
      }
    }
  }
}

// ---------------- Flash-style attention, one wave per (b, h, 16 queries) ----------------
__global__ __launch_bounds__(32) void attn_kernel(const float* __restrict__ Q,
                                                  const float* __restrict__ K,
                                                  const float* __restrict__ V,
                                                  float* __restrict__ O) {
  int qt = blockIdx.x, h = blockIdx.y, b = blockIdx.z, lane = threadIdx.x;
  __shared__ __bf16 ptile[16 * 32];
  const float scale = 0.03608439182435161f;   // 768^-0.5
  const v8f Z8 = {0.f,0.f,0.f,0.f,0.f,0.f,0.f,0.f};
  int n = lane & 15;
  int half16 = lane >> 4;
  int koff = half16 * 16;
  int kbhalf = half16 * 16;
  int q = qt * 16 + n; if (q >= SEQ_) q = SEQ_ - 1;
  const float* qrow = Q + ((size_t)(b * SEQ_ + q)) * D_ + h * HD_;
  v16bf a0, a1;
#pragma unroll
  for (int j = 0; j < 8; ++j) {
    a0[j]     = f2bf(qrow[koff + j]);
    a0[8 + j] = f2bf(qrow[8 + koff + j]);
    a1[j]     = f2bf(qrow[32 + koff + j]);
    a1[8 + j] = f2bf(qrow[40 + koff + j]);
  }
  float mrow[8], lrow[8];
#pragma unroll
  for (int r = 0; r < 8; ++r) { mrow[r] = -3.0e38f; lrow[r] = 0.f; }
  v8f o0 = Z8, o1 = Z8, o2 = Z8, o3 = Z8;

  for (int kb = 0; kb < (SEQ_ + 31) / 32; ++kb) {
    int key0 = kb * 32;
    int keyA = key0 + n;      int keyAc = keyA < SEQ_ ? keyA : SEQ_ - 1;
    int keyB = key0 + 16 + n; int keyBc = keyB < SEQ_ ? keyB : SEQ_ - 1;
    const float* krA = K + ((size_t)(b * SEQ_ + keyAc)) * D_ + h * HD_;
    const float* krB = K + ((size_t)(b * SEQ_ + keyBc)) * D_ + h * HD_;
    v16bf bk0A, bk1A, bk0B, bk1B;
#pragma unroll
    for (int e = 0; e < 16; ++e) {
      bk0A[e] = f2bf(krA[kbhalf + e]);      bk1A[e] = f2bf(krA[32 + kbhalf + e]);
      bk0B[e] = f2bf(krB[kbhalf + e]);      bk1B[e] = f2bf(krB[32 + kbhalf + e]);
    }
    v8f sA = Z8, sB = Z8;
    sA = wmma_bf16(a0, bk0A, sA); sA = wmma_bf16(a1, bk1A, sA);
    sB = wmma_bf16(a0, bk0B, sB); sB = wmma_bf16(a1, bk1B, sB);

    bool vA = keyA < SEQ_, vB = keyB < SEQ_;
#pragma unroll
    for (int r = 0; r < 8; ++r) {
      float sa = vA ? sA[r] * scale : -3.0e38f;
      float sb = vB ? sB[r] * scale : -3.0e38f;
      float vmax = fmaxf(sa, sb);
      for (int o = 8; o >= 1; o >>= 1) vmax = fmaxf(vmax, __shfl_xor(vmax, o, 32));
      float mnew  = fmaxf(mrow[r], vmax);
      float alpha = __expf(mrow[r] - mnew);
      float pa = vA ? __expf(sa - mnew) : 0.f;
      float pb = vB ? __expf(sb - mnew) : 0.f;
      float ps = pa + pb;
      for (int o = 8; o >= 1; o >>= 1) ps += __shfl_xor(ps, o, 32);
      lrow[r] = lrow[r] * alpha + ps;
      mrow[r] = mnew;
      o0[r] = o0[r] * alpha; o1[r] = o1[r] * alpha;
      o2[r] = o2[r] * alpha; o3[r] = o3[r] * alpha;
      int mr = half16 * 8 + r;                 // row of P this lane owns
      ptile[mr * 32 + n]      = f2bf(pa);
      ptile[mr * 32 + 16 + n] = f2bf(pb);
    }
    __syncthreads();
    v16bf pfr;                                  // P rebuilt in A-fragment layout
    const __bf16* prow = &ptile[n * 32];        // row = lane&15
#pragma unroll
    for (int j = 0; j < 8; ++j) { pfr[j] = prow[koff + j]; pfr[8 + j] = prow[8 + koff + j]; }
    __syncthreads();
#pragma unroll
    for (int t = 0; t < 4; ++t) {
      v16bf bv;
#pragma unroll
      for (int e = 0; e < 16; ++e) {
        int key = key0 + kbhalf + e; if (key >= SEQ_) key = SEQ_ - 1;  // P==0 there
        bv[e] = f2bf(V[((size_t)(b * SEQ_ + key)) * D_ + h * HD_ + t * 16 + n]);
      }
      if      (t == 0) o0 = wmma_bf16(pfr, bv, o0);
      else if (t == 1) o1 = wmma_bf16(pfr, bv, o1);
      else if (t == 2) o2 = wmma_bf16(pfr, bv, o2);
      else             o3 = wmma_bf16(pfr, bv, o3);
    }
  }
#pragma unroll
  for (int r = 0; r < 8; ++r) {
    int mq = qt * 16 + half16 * 8 + r;
    if (mq < SEQ_) {
      float invl = 1.f / lrow[r];
      size_t base = ((size_t)(b * SEQ_ + mq)) * D_ + h * HD_;
      O[base + n]      = o0[r] * invl;
      O[base + 16 + n] = o1[r] * invl;
      O[base + 32 + n] = o2[r] * invl;
      O[base + 48 + n] = o3[r] * invl;
    }
  }
}

// ---------------- LayerNorm over rows of 768 ----------------
__global__ __launch_bounds__(256) void ln_kernel(const float* __restrict__ x,
                                                 const float* __restrict__ g,
                                                 const float* __restrict__ bb,
                                                 float* __restrict__ y) {
  int row = blockIdx.x, tid = threadIdx.x;
  const float* xr = x + (size_t)row * D_;
  float a = xr[tid], b2 = xr[tid + 256], c = xr[tid + 512];
  __shared__ float red[256];
  red[tid] = a + b2 + c; __syncthreads();
  for (int o = 128; o > 0; o >>= 1) { if (tid < o) red[tid] += red[tid + o]; __syncthreads(); }
  float mean = red[0] / 768.f; __syncthreads();
  float da = a - mean, db = b2 - mean, dc = c - mean;
  red[tid] = da * da + db * db + dc * dc; __syncthreads();
  for (int o = 128; o > 0; o >>= 1) { if (tid < o) red[tid] += red[tid + o]; __syncthreads(); }
  float inv = rsqrtf(red[0] / 768.f + EPS_);
  float* yr = y + (size_t)row * D_;
  yr[tid]       = da * inv * g[tid]       + bb[tid];
  yr[tid + 256] = db * inv * g[tid + 256] + bb[tid + 256];
  yr[tid + 512] = dc * inv * g[tid + 512] + bb[tid + 512];
}

// ---------------- embedding helpers ----------------
__global__ void patchify_obs_kernel(const float* __restrict__ img, float* __restrict__ outp) {
  int idx = blockIdx.x * blockDim.x + threadIdx.x;
  if (idx >= B_ * 392 * 768) return;
  int f = idx % 768, t = (idx / 768) % 392, b = idx / (768 * 392);
  int s = t & 1, pxy = t >> 1, px = pxy % 14, py = pxy / 14;
  int c = f % 3, jp = (f / 3) % 16, ip = f / 48;
  int row = py * 16 + ip, col = px * 16 + jp;
  outp[idx] = img[(((size_t)b * 224 + row) * 224 + col) * 6 + c * 2 + s];
}
__global__ void patchify_goal_kernel(const float* __restrict__ img, float* __restrict__ outp) {
  int idx = blockIdx.x * blockDim.x + threadIdx.x;
  if (idx >= B_ * 196 * 768) return;
  int f = idx % 768, t = (idx / 768) % 196, b = idx / (768 * 196);
  int px = t % 14, py = t / 14;
  int c = f % 3, jp = (f / 3) % 16, ip = f / 48;
  int row = py * 16 + ip, col = px * 16 + jp;
  outp[idx] = img[(((size_t)b * 224 + row) * 224 + col) * 3 + c];
}
__global__ __launch_bounds__(768) void cls_text_kernel(const float* __restrict__ cls_tok,
                                                       const float* __restrict__ tok_emb,
                                                       const int* __restrict__ txt,
                                                       float* __restrict__ X) {
  int b = blockIdx.y, which = blockIdx.x, d = threadIdx.x;
  if (which == 0) {
    X[((size_t)b * SEQ_) * D_ + d] = cls_tok[d];
  } else {
    int t = which - 1;
    int tok = txt[b * 77 + t];
    X[((size_t)(b * SEQ_ + 394 + t)) * D_ + d] = tok_emb[(size_t)tok * D_ + d];
  }
}
__global__ __launch_bounds__(768) void pose_h_kernel(const float* __restrict__ pose,
                                                     const float* __restrict__ w1,
                                                     const float* __restrict__ b1,
                                                     float* __restrict__ hid) {
  int b = blockIdx.x, j = threadIdx.x;
  float s = b1[j];
  for (int i = 0; i < 7; ++i) s += pose[b * 7 + i] * w1[i * D_ + j];
  hid[b * D_ + j] = fmaxf(s, 0.f);
}
__global__ __launch_bounds__(768) void pose_o_kernel(const float* __restrict__ hid,
                                                     const float* __restrict__ w2,
                                                     const float* __restrict__ b2,
                                                     float* __restrict__ X) {
  int b = blockIdx.x, j = threadIdx.x;
  float s = b2[j];
  for (int i = 0; i < D_; ++i) s += hid[b * D_ + i] * w2[(size_t)i * D_ + j];
  X[((size_t)(b * SEQ_ + 1)) * D_ + j] = s;
}
__global__ void posadd_kernel(float* __restrict__ X) {
  int idx = blockIdx.x * blockDim.x + threadIdx.x;
  if (idx >= B_ * SEQ_ * D_) return;
  int d = idx % D_, t = (idx / D_) % SEQ_;
  int je = d & ~1;
  float freq = __expf(-(float)je * (9.210340371976184f / 768.f));  // 10000^(-je/768)
  float ang = (float)t * freq;
  X[idx] += (d & 1) ? __cosf(ang) : __sinf(ang);
}

// ---------------- final head: hln(LN) + 768x7 matmul ----------------
__global__ __launch_bounds__(256) void head_kernel(const float* __restrict__ Hlnf,
                                                   const float* __restrict__ hg,
                                                   const float* __restrict__ hb,
                                                   const float* __restrict__ hw,
                                                   const float* __restrict__ hbias,
                                                   float* __restrict__ outp) {
  int b = blockIdx.x, tid = threadIdx.x;
  const float* xr = Hlnf + (size_t)(b * SEQ_) * D_;   // cls row (already lnf-normalized)
  __shared__ float red[256];
  __shared__ float xn[768];
  float a = xr[tid], c = xr[tid + 256], e = xr[tid + 512];
  red[tid] = a + c + e; __syncthreads();
  for (int o = 128; o > 0; o >>= 1) { if (tid < o) red[tid] += red[tid + o]; __syncthreads(); }
  float mean = red[0] / 768.f; __syncthreads();
  float da = a - mean, dc = c - mean, de = e - mean;
  red[tid] = da * da + dc * dc + de * de; __syncthreads();
  for (int o = 128; o > 0; o >>= 1) { if (tid < o) red[tid] += red[tid + o]; __syncthreads(); }
  float inv = rsqrtf(red[0] / 768.f + EPS_);
  xn[tid]       = da * inv * hg[tid]       + hb[tid];
  xn[tid + 256] = dc * inv * hg[tid + 256] + hb[tid + 256];
  xn[tid + 512] = de * inv * hg[tid + 512] + hb[tid + 512];
  __syncthreads();
  if (tid < 7) {
    float s = hbias[tid];
    for (int dd = 0; dd < 768; ++dd) s += xn[dd] * hw[dd * 7 + tid];
    outp[b * 7 + tid] = s;
  }
}

// ---------------- host orchestration ----------------
extern "C" void kernel_launch(void* const* d_in, const int* in_sizes, int n_in,
                              void* d_out, int out_size, void* d_ws, size_t ws_size,
                              hipStream_t stream) {
  const float* images    = (const float*)d_in[0];
  const float* goal_imgs = (const float*)d_in[1];
  const float* pose      = (const float*)d_in[2];
  const int*   goals_txt = (const int*)  d_in[3];
  const float* obs_w  = (const float*)d_in[4];
  const float* obs_b  = (const float*)d_in[5];
  const float* goal_w = (const float*)d_in[6];
  const float* goal_b = (const float*)d_in[7];
  const float* tok_emb = (const float*)d_in[8];
  const float* pose_w1 = (const float*)d_in[9];
  const float* pose_b1 = (const float*)d_in[10];
  const float* pose_w2 = (const float*)d_in[11];
  const float* pose_b2 = (const float*)d_in[12];
  const float* cls_tok = (const float*)d_in[13];
  const float* wq = (const float*)d_in[14];
  const float* wk = (const float*)d_in[15];
  const float* wv = (const float*)d_in[16];
  const float* proj_w = (const float*)d_in[17];
  const float* proj_b = (const float*)d_in[18];
  const float* ln1_g = (const float*)d_in[19];
  const float* ln1_b = (const float*)d_in[20];
  const float* ln2_g = (const float*)d_in[21];
  const float* ln2_b = (const float*)d_in[22];
  const float* ff_w1 = (const float*)d_in[23];
  const float* ff_b1 = (const float*)d_in[24];
  const float* ff_w2 = (const float*)d_in[25];
  const float* ff_b2 = (const float*)d_in[26];
  const float* lnf_g = (const float*)d_in[27];
  const float* lnf_b = (const float*)d_in[28];
  const float* hln_g = (const float*)d_in[29];
  const float* hln_b = (const float*)d_in[30];
  const float* head_w = (const float*)d_in[31];
  const float* head_b = (const float*)d_in[32];
  float* outp = (float*)d_out;

  // workspace layout (floats): X | H | Q | K | V | O | Apack(bf16) | Bpack(bf16) | poseH
  const size_t AF = (size_t)MTOK * D_;             // 4,098,048 floats
  float* X  = (float*)d_ws;
  float* Hb = X + AF;
  float* Qb = X + 2 * AF;
  float* Kb = X + 3 * AF;
  float* Vb = X + 4 * AF;
  float* Ob = X + 5 * AF;
  float* F1 = Qb;                                   // FFN hidden aliases Q..O (exact fit)
  __bf16* Ap = (__bf16*)(X + 6 * AF);
  const size_t APACK_ELEMS = 334ull * 96 * 512;     // Mtiles_max * Ktiles_max * 512
  __bf16* Bp = Ap + APACK_ELEMS;
  const size_t BPACK_ELEMS = 96ull * 48 * 512;      // == 24*192*512
  float* poseH = (float*)(Bp + BPACK_ELEMS);
  float* Pobs  = Qb;                                // 3136x768 staging (pre-layer only)
  float* Pgoal = Vb;                                // 1568x768 staging

  // ---- embeddings ----
  { int tot = B_ * 392 * 768; patchify_obs_kernel <<<(tot + 255) / 256, 256, 0, stream>>>(images, Pobs); }
  { int tot = B_ * 196 * 768; patchify_goal_kernel<<<(tot + 255) / 256, 256, 0, stream>>>(goal_imgs, Pgoal); }

  pack_a_kernel<<<dim3(24, 196), 32, 0, stream>>>(Pobs, Ap, 3136, 768);
  pack_b_kernel<<<dim3(48, 24),  32, 0, stream>>>(obs_w, Bp, 768, 768);
  gemm_wmma_kernel<<<dim3(12, 25), 128, 0, stream>>>(Ap, Bp, obs_b, nullptr, X,
                                                     3136, 768, 24, 392, 2, SEQ_, 0);
  pack_a_kernel<<<dim3(24, 98), 32, 0, stream>>>(Pgoal, Ap, 1568, 768);
  pack_b_kernel<<<dim3(48, 24), 32, 0, stream>>>(goal_w, Bp, 768, 768);
  gemm_wmma_kernel<<<dim3(12, 13), 128, 0, stream>>>(Ap, Bp, goal_b, nullptr, X,
                                                     1568, 768, 24, 196, 471, SEQ_, 0);
  pose_h_kernel<<<8, 768, 0, stream>>>(pose, pose_w1, pose_b1, poseH);
  pose_o_kernel<<<8, 768, 0, stream>>>(poseH, pose_w2, pose_b2, X);
  cls_text_kernel<<<dim3(78, 8), 768, 0, stream>>>(cls_tok, tok_emb, goals_txt, X);
  { int tot = B_ * SEQ_ * D_; posadd_kernel<<<(tot + 255) / 256, 256, 0, stream>>>(X); }

  // ---- transformer layers ----
  const int Mt  = (MTOK + 15) / 16;   // 334 M-tiles
  const int Mg  = (Mt + 7) / 8;       // 42 gemm row-blocks (128 rows each)
  for (int l = 0; l < 12; ++l) {
    const float* wq_l = wq + (size_t)l * D_ * D_;
    const float* wk_l = wk + (size_t)l * D_ * D_;
    const float* wv_l = wv + (size_t)l * D_ * D_;
    const float* pw_l = proj_w + (size_t)l * D_ * D_;
    const float* pb_l = proj_b + (size_t)l * D_;
    const float* f1w  = ff_w1 + (size_t)l * D_ * FF_;
    const float* f1b  = ff_b1 + (size_t)l * FF_;
    const float* f2w  = ff_w2 + (size_t)l * FF_ * D_;
    const float* f2b  = ff_b2 + (size_t)l * D_;

    ln_kernel<<<MTOK, 256, 0, stream>>>(X, ln1_g + l * D_, ln1_b + l * D_, Hb);
    pack_a_kernel<<<dim3(24, Mt), 32, 0, stream>>>(Hb, Ap, MTOK, D_);
    pack_b_kernel<<<dim3(48, 24), 32, 0, stream>>>(wq_l, Bp, D_, D_);
    gemm_wmma_kernel<<<dim3(12, Mg), 128, 0, stream>>>(Ap, Bp, nullptr, nullptr, Qb,
                                                       MTOK, D_, 24, MTOK, 0, 0, 0);
    pack_b_kernel<<<dim3(48, 24), 32, 0, stream>>>(wk_l, Bp, D_, D_);
    gemm_wmma_kernel<<<dim3(12, Mg), 128, 0, stream>>>(Ap, Bp, nullptr, nullptr, Kb,
                                                       MTOK, D_, 24, MTOK, 0, 0, 0);
    pack_b_kernel<<<dim3(48, 24), 32, 0, stream>>>(wv_l, Bp, D_, D_);
    gemm_wmma_kernel<<<dim3(12, Mg), 128, 0, stream>>>(Ap, Bp, nullptr, nullptr, Vb,
                                                       MTOK, D_, 24, MTOK, 0, 0, 0);

    attn_kernel<<<dim3(42, H_, B_), 32, 0, stream>>>(Qb, Kb, Vb, Ob);

    pack_a_kernel<<<dim3(24, Mt), 32, 0, stream>>>(Ob, Ap, MTOK, D_);
    pack_b_kernel<<<dim3(48, 24), 32, 0, stream>>>(pw_l, Bp, D_, D_);
    gemm_wmma_kernel<<<dim3(12, Mg), 128, 0, stream>>>(Ap, Bp, pb_l, X, X,
                                                       MTOK, D_, 24, MTOK, 0, 0, 0);

    ln_kernel<<<MTOK, 256, 0, stream>>>(X, ln2_g + l * D_, ln2_b + l * D_, Hb);
    pack_a_kernel<<<dim3(24, Mt), 32, 0, stream>>>(Hb, Ap, MTOK, D_);
    pack_b_kernel<<<dim3(192, 24), 32, 0, stream>>>(f1w, Bp, D_, FF_);
    gemm_wmma_kernel<<<dim3(48, Mg), 128, 0, stream>>>(Ap, Bp, f1b, nullptr, F1,
                                                       MTOK, FF_, 24, MTOK, 0, 0, 1);
    pack_a_kernel<<<dim3(96, Mt), 32, 0, stream>>>(F1, Ap, MTOK, FF_);
    pack_b_kernel<<<dim3(48, 96), 32, 0, stream>>>(f2w, Bp, FF_, D_);
    gemm_wmma_kernel<<<dim3(12, Mg), 128, 0, stream>>>(Ap, Bp, f2b, X, X,
                                                       MTOK, D_, 96, MTOK, 0, 0, 0);
  }

  // ---- final LN + head ----
  ln_kernel<<<MTOK, 256, 0, stream>>>(X, lnf_g, lnf_b, Hb);
  head_kernel<<<8, 256, 0, stream>>>(Hb, hln_g, hln_b, head_w, head_b, outp);
}
